// MultiHeadAttention_36575941493496
// MI455X (gfx1250) — matrix operand order repaired and checked by reading, not compiled, not run on hardware
//
#include <hip/hip_runtime.h>
#include <stdint.h>

// ---------------------------------------------------------------------------
// MHA forward for MI455X (gfx1250, wave32, WMMA bf16 16x16x32, f32 accum)
// B=2, S=2048, D=1024, H=16, DK=DV=64
// GEMM weight tiles staged to LDS via Tensor Data Mover (tensor_load_to_lds)
// ---------------------------------------------------------------------------

typedef __bf16 bf16;
typedef __attribute__((ext_vector_type(16))) __bf16 v16bf;
typedef __attribute__((ext_vector_type(8)))  float  v8f;
typedef __attribute__((ext_vector_type(4)))  unsigned int v4u;
typedef __attribute__((ext_vector_type(8)))  int v8i;
typedef __attribute__((ext_vector_type(4)))  int v4i;

#if __has_builtin(__builtin_amdgcn_tensor_load_to_lds)
#define USE_TDM 1
#else
#define USE_TDM 0
#endif

union U16 {
    v16bf v;
    uint4 q[2];
};
static_assert(sizeof(U16) == 32, "U16 must be 32 bytes");

static __device__ __forceinline__ v8f wmma_bf16(const U16& a, const U16& b, v8f c) {
    return __builtin_amdgcn_wmma_f32_16x16x32_bf16(
        /*neg_a=*/false, a.v, /*neg_b=*/false, b.v,
        /*c_mod=*/(short)0, c, /*reuse_a=*/false, /*reuse_b=*/false);
}

static __device__ __forceinline__ v8f zero8() {
    v8f z = {0.f, 0.f, 0.f, 0.f, 0.f, 0.f, 0.f, 0.f};
    return z;
}

// ---------------------------------------------------------------------------
// TDM: DMA a [tile_d1 rows x tile_d0 elems] bf16 tile (row stride
// `row_stride_elems`) from global into LDS, padding each 32-DWORD row with
// 4 DWORDs so LDS row stride = 72 halfs (conflict-free for fragment reads).
// D# layout per CDNA5 ISA; 2-D => groups 2,3 = 0.
// This toolchain's builtin takes 6 args (g0, g1, g2, g3, extra, cpol).
// ---------------------------------------------------------------------------
#if USE_TDM
static __device__ __forceinline__ void tdm_load_tile_bf16(
    const bf16* gsrc, unsigned lds_byte_addr,
    unsigned tile_d0, unsigned tile_d1, unsigned row_stride_elems) {
    const unsigned long long ga = (unsigned long long)(uintptr_t)gsrc;
    v4u g0;
    g0[0] = 1u;                                   // count=1, user mode
    g0[1] = lds_byte_addr;                        // lds_addr
    g0[2] = (unsigned)(ga & 0xFFFFFFFFu);         // global_addr[31:0]
    g0[3] = (unsigned)((ga >> 32) & 0x1FFFFFFu)   // global_addr[56:32]
          | (2u << 30);                           // type = 2 ("image")
    const unsigned td0 = 1u << 20, td1 = 1u << 20;  // huge dims: no OOB clip
    v8i g1;
    g1[0] = (int)((1u << 16)                      // data_size = 2 bytes
                | (1u << 20)                      // pad_enable
                | (4u << 22)                      // pad_interval: 32 DWORDs
                | (3u << 25));                    // pad_amount:   4 DWORDs
    g1[1] = (int)((td0 & 0xFFFFu) << 16);         // atomic_barrier=0 | td0.lo
    g1[2] = (int)((td0 >> 16) | ((td1 & 0xFFFFu) << 16));
    g1[3] = (int)((td1 >> 16) | (tile_d0 << 16)); // td1.hi | tile_dim0
    g1[4] = (int)tile_d1;                         // tile_dim1 | tile_dim2=0
    g1[5] = (int)row_stride_elems;                // tensor_dim0_stride[31:0]
    g1[6] = 0;                                    // stride0.hi | stride1.lo
    g1[7] = 0;
    const v4i z4 = {0, 0, 0, 0};
    const v8i z8 = {0, 0, 0, 0, 0, 0, 0, 0};
    __builtin_amdgcn_tensor_load_to_lds(g0, g1, z4, z4, z8, 0);
}
#endif

static __device__ __forceinline__ void wait_tensorcnt0() {
#if __has_builtin(__builtin_amdgcn_s_wait_tensorcnt)
    __builtin_amdgcn_s_wait_tensorcnt(0);
#else
    asm volatile("s_wait_tensorcnt 0x0" ::: "memory");
#endif
}

// ---------------------------------------------------------------------------
// fp32 -> bf16 conversion (weights)
// ---------------------------------------------------------------------------
__global__ void f32_to_bf16_kernel(const float* __restrict__ src,
                                   bf16* __restrict__ dst, int n) {
    int i = (blockIdx.x * blockDim.x + threadIdx.x) * 4;
    if (i + 3 < n) {
        float4 f = *reinterpret_cast<const float4*>(src + i);
        dst[i + 0] = (bf16)f.x;
        dst[i + 1] = (bf16)f.y;
        dst[i + 2] = (bf16)f.z;
        dst[i + 3] = (bf16)f.w;
    }
}

// ---------------------------------------------------------------------------
// Tiled GEMM:  C[M,N] = A[M,Kd] * W[N,Kd]^T + bias[N]
//   A fp32 (VALU-converted to bf16 into LDS), W bf16 staged by TDM
//   (double-buffered: DMA of tile k+1 overlaps WMMA on tile k).
//   Block tile 128x128, K-step 64, 256 threads = 8 waves.
// MODE 0: out bf16 [B,H,S,64] (q/k) | MODE 1: bf16 [B,H,64,S] (v^T)
// MODE 2: out fp32 row-major [M,N]
// ---------------------------------------------------------------------------
#define TM 128
#define TN 128
#define TK 64
#define LDH 72   // halfs per LDS row: 144B stride, 16B aligned, conflict-free

template <int MODE>
__global__ __launch_bounds__(256) void gemm_bf16_kernel(
    const float* __restrict__ A, const bf16* __restrict__ W,
    const float* __restrict__ bias, void* __restrict__ out,
    int M, int N, int Kd) {
    __shared__ bf16 sA[TM * LDH];
    __shared__ bf16 sB[2][TN * LDH];

    const int t    = threadIdx.x;
    const int w    = t >> 5;
    const int lane = t & 31;
    const int lrow = lane & 15;
    const int lhi  = lane >> 4;
    const int wm   = w & 1;   // 2 wave-rows of 64
    const int wn   = w >> 1;  // 4 wave-cols of 32
    const int m0   = blockIdx.x * TM;
    const int n0   = blockIdx.y * TN;

    v8f acc[4][2];
#pragma unroll
    for (int mt = 0; mt < 4; ++mt)
#pragma unroll
        for (int nt = 0; nt < 2; ++nt) acc[mt][nt] = zero8();

    const int r  = t >> 1;
    const int c0 = (t & 1) * 32;

#if USE_TDM
    if (t < 32) {  // wave 0 issues the first B-tile DMA
        tdm_load_tile_bf16(&W[(size_t)n0 * Kd], (unsigned)(uintptr_t)&sB[0][0],
                           TK, TN, Kd);
    }
#endif

    int p = 0;
    for (int kb = 0; kb < Kd; kb += TK) {
        __syncthreads();  // prior compute done before LDS overwrite
        // ---- stage A tile (fp32 -> bf16): 128 rows x 64 cols
        {
            const float4* g =
                reinterpret_cast<const float4*>(&A[(size_t)(m0 + r) * Kd + kb + c0]);
            if (kb + TK < Kd)
                __builtin_prefetch(&A[(size_t)(m0 + r) * Kd + kb + TK + c0], 0, 1);
            bf16* d = &sA[r * LDH + c0];
#pragma unroll
            for (int j = 0; j < 8; ++j) {
                float4 f = g[j];
                d[4 * j + 0] = (bf16)f.x;
                d[4 * j + 1] = (bf16)f.y;
                d[4 * j + 2] = (bf16)f.z;
                d[4 * j + 3] = (bf16)f.w;
            }
        }
#if USE_TDM
        if (t < 32) {
            wait_tensorcnt0();  // sB[p] fully landed
            if (kb + TK < Kd)   // kick DMA for next tile into the other buffer
                tdm_load_tile_bf16(&W[(size_t)n0 * Kd + kb + TK],
                                   (unsigned)(uintptr_t)&sB[p ^ 1][0], TK, TN, Kd);
        }
#else
        {
            const uint4* g =
                reinterpret_cast<const uint4*>(&W[(size_t)(n0 + r) * Kd + kb + c0]);
            uint4* d = reinterpret_cast<uint4*>(&sB[p][r * LDH + c0]);
#pragma unroll
            for (int j = 0; j < 4; ++j) d[j] = g[j];
        }
#endif
        __syncthreads();

#pragma unroll
        for (int ks = 0; ks < TK; ks += 32) {
            // A fragments for 4 M-tiles (16-bit A 16x32 layout, ISA 7.12.2)
            U16 a[4];
#pragma unroll
            for (int mt = 0; mt < 4; ++mt) {
                const int row = wm * 64 + mt * 16 + lrow;
                const bf16* base = &sA[row * LDH + ks + lhi * 8];
                a[mt].q[0] = *reinterpret_cast<const uint4*>(base);
                a[mt].q[1] = *reinterpret_cast<const uint4*>(base + 16);
            }
#pragma unroll
            for (int nt = 0; nt < 2; ++nt) {
                const int col = wn * 32 + nt * 16 + lrow;
                const bf16* base = &sB[p][col * LDH + ks + lhi * 8];
                U16 b;
                b.q[0] = *reinterpret_cast<const uint4*>(base);
                b.q[1] = *reinterpret_cast<const uint4*>(base + 16);
#pragma unroll
                for (int mt = 0; mt < 4; ++mt)
                    acc[mt][nt] = wmma_bf16(a[mt], b, acc[mt][nt]);
            }
        }
        p ^= 1;
    }

    // ---- epilogue: C layout: VGPR v, lanes 0-15 row=v, lanes 16-31 row=v+8
#pragma unroll
    for (int nt = 0; nt < 2; ++nt) {
        const int gn   = n0 + wn * 32 + nt * 16 + lrow;
        const float bv = bias[gn];
#pragma unroll
        for (int mt = 0; mt < 4; ++mt) {
#pragma unroll
            for (int v = 0; v < 8; ++v) {
                const int gm  = m0 + wm * 64 + mt * 16 + v + 8 * lhi;
                const float x = acc[mt][nt][v] + bv;
                if (MODE == 0) {
                    const int b = gm >> 11, s = gm & 2047;
                    const int h = gn >> 6, d = gn & 63;
                    ((bf16*)out)[(((size_t)b * 16 + h) * 2048 + s) * 64 + d] = (bf16)x;
                } else if (MODE == 1) {
                    const int b = gm >> 11, s = gm & 2047;
                    const int h = gn >> 6, d = gn & 63;
                    ((bf16*)out)[(((size_t)b * 16 + h) * 64 + d) * 2048 + s] = (bf16)x;
                } else {
                    ((float*)out)[(size_t)gm * N + gn] = x;
                }
            }
        }
    }
}

// ---------------------------------------------------------------------------
// Flash attention: grid (S/64, B*H), block 128 (4 waves).
// Wave w owns query rows [qt*64 + 16w, +16) and the full 64-wide key tiles.
// q,k: bf16 [B,H,S,64]; vt: bf16 [B,H,64,S]; ctx out: fp32 [B,S,H*64].
// ---------------------------------------------------------------------------
__global__ __launch_bounds__(128) void attn_kernel(
    const bf16* __restrict__ q, const bf16* __restrict__ k,
    const bf16* __restrict__ vt, const unsigned char* __restrict__ maskp,
    float* __restrict__ ctx) {
    __shared__ bf16 sP[4][16 * LDH];  // per-wave P tile (16 rows x 64 keys)

    const int t    = threadIdx.x;
    const int w    = t >> 5;
    const int lane = t & 31;
    const int lrow = lane & 15;
    const int lhi  = lane >> 4;
    const int qt   = blockIdx.x;  // query tile (64 rows)
    const int bh   = blockIdx.y;  // b*16 + h
    const int b    = bh >> 4;
    const int h    = bh & 15;

    const size_t base  = (size_t)bh * 2048 * 64;  // q/k head base (halfs)
    const size_t vbase = (size_t)bh * 64 * 2048;  // vt head base (halfs)
    const int mrow0 = qt * 64 + w * 16;

    // ---- Q fragments for both 32-deep K-steps (loaded once)
    U16 a0, a1;
    {
        const bf16* qp = q + base + (size_t)(mrow0 + lrow) * 64 + lhi * 8;
        const uint4* p = reinterpret_cast<const uint4*>(qp);
        a0.q[0] = p[0];  // halfs +0..7
        a0.q[1] = p[2];  // halfs +16..23
        a1.q[0] = p[4];  // halfs +32..39
        a1.q[1] = p[6];  // halfs +48..55
    }

    v8f O[4] = {zero8(), zero8(), zero8(), zero8()};
    float mrow[8], lsum[8];
#pragma unroll
    for (int v = 0; v < 8; ++v) { mrow[v] = -1e30f; lsum[v] = 0.f; }

    const float scale = 0.125f;  // 1/sqrt(64)

    for (int kt = 0; kt < 2048; kt += 64) {
        // ---- scores tile: 16 rows x 64 keys (4 N-tiles)
        v8f s[4];
#pragma unroll
        for (int nt = 0; nt < 4; ++nt) {
            const bf16* kp = k + base + (size_t)(kt + nt * 16 + lrow) * 64;
            U16 b0, b1;
            {
                const uint4* p = reinterpret_cast<const uint4*>(kp + lhi * 8);
                b0.q[0] = p[0];
                b0.q[1] = p[2];
                b1.q[0] = p[4];
                b1.q[1] = p[6];
            }
            v8f z = zero8();
            z = wmma_bf16(a0, b0, z);
            s[nt] = wmma_bf16(a1, b1, z);
        }
        // ---- scale + key-padding mask (column-uniform per lane)
#pragma unroll
        for (int nt = 0; nt < 4; ++nt) {
            const bool masked = maskp[(size_t)b * 2048 + kt + nt * 16 + lrow] != 0;
#pragma unroll
            for (int v = 0; v < 8; ++v) {
                const float sv = s[nt][v] * scale;
                s[nt][v] = masked ? -1e30f : sv;
            }
        }
        // ---- online softmax (per-row butterflies within 16-lane halves)
#pragma unroll
        for (int v = 0; v < 8; ++v) {
            float tmax = fmaxf(fmaxf(s[0][v], s[1][v]), fmaxf(s[2][v], s[3][v]));
#pragma unroll
            for (int m = 8; m >= 1; m >>= 1)
                tmax = fmaxf(tmax, __shfl_xor(tmax, m, 32));
            const float mnew = fmaxf(mrow[v], tmax);
            const float corr = __expf(mrow[v] - mnew);
            float psum = 0.f;
#pragma unroll
            for (int nt = 0; nt < 4; ++nt) {
                const float pv = __expf(s[nt][v] - mnew);
                s[nt][v] = pv;
                psum += pv;
            }
#pragma unroll
            for (int m = 8; m >= 1; m >>= 1)
                psum += __shfl_xor(psum, m, 32);
            lsum[v] = lsum[v] * corr + psum;
            mrow[v] = mnew;
#pragma unroll
            for (int ot = 0; ot < 4; ++ot) O[ot][v] *= corr;
        }
        // ---- P: C-layout -> A-fragment layout via per-wave LDS tile
        bf16* pw = &sP[w][0];
#pragma unroll
        for (int nt = 0; nt < 4; ++nt)
#pragma unroll
            for (int v = 0; v < 8; ++v)
                pw[(v + 8 * lhi) * LDH + nt * 16 + lrow] = (bf16)s[nt][v];
        asm volatile("s_wait_dscnt 0x0" ::: "memory");  // LDS in-order per wave
        U16 pa0, pa1;
        {
            const bf16* pr = pw + lrow * LDH + lhi * 8;
            const uint4* p = reinterpret_cast<const uint4*>(pr);
            pa0.q[0] = p[0];
            pa0.q[1] = p[2];
            pa1.q[0] = p[4];
            pa1.q[1] = p[6];
        }
        // ---- O += P @ V  (V pre-transposed: contiguous B-fragment loads)
#pragma unroll
        for (int ot = 0; ot < 4; ++ot) {
            const bf16* vp = vt + vbase + (size_t)(ot * 16 + lrow) * 2048 + kt;
            U16 vb0, vb1;
            {
                const uint4* p = reinterpret_cast<const uint4*>(vp + lhi * 8);
                vb0.q[0] = p[0];
                vb0.q[1] = p[2];
                vb1.q[0] = p[4];
                vb1.q[1] = p[6];
            }
            O[ot] = wmma_bf16(pa0, vb0, O[ot]);
            O[ot] = wmma_bf16(pa1, vb1, O[ot]);
        }
    }

    // ---- normalize + write ctx fp32 [B,S,H*64]
#pragma unroll
    for (int v = 0; v < 8; ++v) {
        const float inv = 1.0f / lsum[v];
        const int row = mrow0 + v + 8 * lhi;
#pragma unroll
        for (int ot = 0; ot < 4; ++ot) {
            const int col = h * 64 + ot * 16 + lrow;
            ctx[((size_t)b * 2048 + row) * 1024 + col] = O[ot][v] * inv;
        }
    }
}

// ---------------------------------------------------------------------------
// Host launcher
// ---------------------------------------------------------------------------
extern "C" void kernel_launch(void* const* d_in, const int* in_sizes, int n_in,
                              void* d_out, int out_size, void* d_ws, size_t ws_size,
                              hipStream_t stream) {
    (void)in_sizes; (void)n_in; (void)out_size; (void)ws_size;

    const float* Q  = (const float*)d_in[0];
    const float* K  = (const float*)d_in[1];
    const float* V  = (const float*)d_in[2];
    const unsigned char* mask = (const unsigned char*)d_in[3];
    const float* Wq = (const float*)d_in[4];
    const float* bq = (const float*)d_in[5];
    const float* Wk = (const float*)d_in[6];
    const float* bk = (const float*)d_in[7];
    const float* Wv = (const float*)d_in[8];
    const float* bv = (const float*)d_in[9];
    const float* Wo = (const float*)d_in[10];
    const float* bo = (const float*)d_in[11];

    char* ws = (char*)d_ws;
    const size_t MB = 1u << 20;
    bf16* wq_b = (bf16*)(ws + 0 * MB);   // 2 MB each
    bf16* wk_b = (bf16*)(ws + 2 * MB);
    bf16* wv_b = (bf16*)(ws + 4 * MB);
    bf16* wo_b = (bf16*)(ws + 6 * MB);
    bf16* qh   = (bf16*)(ws + 8 * MB);   // [B,H,S,64] bf16, 8 MB
    bf16* kh   = (bf16*)(ws + 16 * MB);  // [B,H,S,64] bf16, 8 MB
    bf16* vth  = (bf16*)(ws + 24 * MB);  // [B,H,64,S] bf16, 8 MB
    float* ctx = (float*)(ws + 32 * MB); // [B,S,1024] f32, 16 MB

    const int WN = 1 << 20;  // 1024*1024 weight elems
    f32_to_bf16_kernel<<<WN / 1024, 256, 0, stream>>>(Wq, wq_b, WN);
    f32_to_bf16_kernel<<<WN / 1024, 256, 0, stream>>>(Wk, wk_b, WN);
    f32_to_bf16_kernel<<<WN / 1024, 256, 0, stream>>>(Wv, wv_b, WN);
    f32_to_bf16_kernel<<<WN / 1024, 256, 0, stream>>>(Wo, wo_b, WN);

    const dim3 gg(4096 / TM, 1024 / TN);  // (32, 8)
    gemm_bf16_kernel<0><<<gg, 256, 0, stream>>>(Q, wq_b, bq, qh, 4096, 1024, 1024);
    gemm_bf16_kernel<0><<<gg, 256, 0, stream>>>(K, wk_b, bk, kh, 4096, 1024, 1024);
    gemm_bf16_kernel<1><<<gg, 256, 0, stream>>>(V, wv_b, bv, vth, 4096, 1024, 1024);

    attn_kernel<<<dim3(2048 / 64, 32), 128, 0, stream>>>(qh, kh, vth, mask, ctx);

    gemm_bf16_kernel<2><<<gg, 256, 0, stream>>>(ctx, wo_b, bo, d_out, 4096, 1024, 1024);
}
